// PatternCreativityAttention_8358006358353
// MI455X (gfx1250) — compile-verified, matrix operand order checked by reading, not run on hardware
//
#include <hip/hip_runtime.h>
#include <hip/hip_bf16.h>

// ---------------------------------------------------------------------------
// CDNA5 (gfx1250) WMMA attention implementation.
// wave32; V_WMMA_F32_16X16X32_BF16 for all GEMMs, fp32 softmax/LN.
// ---------------------------------------------------------------------------

typedef __attribute__((ext_vector_type(16))) __bf16 v16bf;
typedef __attribute__((ext_vector_type(8)))  __bf16 v8bf;
typedef __attribute__((ext_vector_type(8)))  float  v8f;

// A-like fragment load: 16x32 bf16 tile, row-major, row stride `ld` elements.
// Per CDNA5 ISA 16-bit A layout: lane l -> row (l&15); K chunk base (l>>4)*8,
// elements 0..7 cover K=base..base+7, elements 8..15 cover K=base+16..base+23.
// Both A and B operands of C = X * W^T load with this pattern (B^T stored like A).
__device__ __forceinline__ v16bf frag_ld_bf16(const __bf16* __restrict__ p0,
                                              int ld, int lane) {
  const int m = lane & 15;
  const int h = lane >> 4;
  const __bf16* p = p0 + m * ld + h * 8;
  v8bf lo = *reinterpret_cast<const v8bf*>(p);        // global_load_b128
  v8bf hi = *reinterpret_cast<const v8bf*>(p + 16);   // global_load_b128
  v16bf f;
#pragma unroll
  for (int e = 0; e < 8; ++e) { f[e] = lo[e]; f[e + 8] = hi[e]; }
  return f;
}

// Same pattern but source is fp32 in LDS (probabilities), convert to bf16.
__device__ __forceinline__ v16bf frag_ld_f32(const float* __restrict__ p0,
                                             int ld, int lane) {
  const int m = lane & 15;
  const int h = lane >> 4;
  const float* p = p0 + m * ld + h * 8;
  v16bf f;
#pragma unroll
  for (int e = 0; e < 8; ++e) {
    f[e]     = (__bf16)p[e];
    f[e + 8] = (__bf16)p[e + 16];
  }
  return f;
}

__device__ __forceinline__ v8f wmma_bf16(v16bf a, v16bf b, v8f c) {
  return __builtin_amdgcn_wmma_f32_16x16x32_bf16(
      /*neg_a=*/false, a, /*neg_b=*/false, b,
      /*c_mod=*/(short)0, c, /*reuse_a=*/false, /*reuse_b=*/false);
}

// ---------------------------------------------------------------------------
// Kernel 1: diversity bias  div[b,p,q] = (1 - softmax(pf@pf^T))*0.1
// pf: (32, 64, 128) fp32.  One block per batch, 64 threads (one row each).
// ---------------------------------------------------------------------------
__global__ void __launch_bounds__(64)
k_div_bias(const float* __restrict__ pf, float* __restrict__ div) {
  __shared__ float pf_s[64 * 128];
  const int b = blockIdx.x;
  const int tid = threadIdx.x;
  const float* src = pf + (size_t)b * 64 * 128;
  for (int i = tid; i < 64 * 128; i += 64) pf_s[i] = src[i];
  __syncthreads();

  float sim[64];
  const float* rp = pf_s + tid * 128;
#pragma unroll 4
  for (int q = 0; q < 64; ++q) {
    const float* qp = pf_s + q * 128;
    float s = 0.f;
    for (int f = 0; f < 128; ++f) s = fmaf(rp[f], qp[f], s);
    sim[q] = s;
  }
  float mx = sim[0];
  for (int q = 1; q < 64; ++q) mx = fmaxf(mx, sim[q]);
  float sum = 0.f;
  for (int q = 0; q < 64; ++q) { sim[q] = __expf(sim[q] - mx); sum += sim[q]; }
  const float inv = 1.f / sum;
  float* dst = div + (size_t)b * 4096 + tid * 64;
  for (int q = 0; q < 64; ++q) dst[q] = (1.f - sim[q] * inv) * 0.1f;
}

// ---------------------------------------------------------------------------
// Kernel 2: fp32 -> bf16 conversion (grid-stride)
// ---------------------------------------------------------------------------
__global__ void k_cvt(const float* __restrict__ s, __bf16* __restrict__ d, int n) {
  int i = blockIdx.x * blockDim.x + threadIdx.x;
  const int stride = gridDim.x * blockDim.x;
  for (; i < n; i += stride) d[i] = (__bf16)s[i];
}

// ---------------------------------------------------------------------------
// Kernel 3: QKV projection  Y = x_seq @ W^T + bias  (8192 x 1024 x 1024)
// grid = (512 strips of 16 rows, 3 projections), block = 256 (8 waves).
// Wave w computes rows [strip*16, +16) x cols [w*128, +128): 8 accumulators.
// q,k stored (B,nh,S,dk) bf16;  v stored transposed (B,nh,dk,S) bf16.
// ---------------------------------------------------------------------------
__global__ void __launch_bounds__(256)
k_qkv(const __bf16* __restrict__ xb,
      const __bf16* __restrict__ wqb, const __bf16* __restrict__ wkb,
      const __bf16* __restrict__ wvb,
      const float* __restrict__ bq, const float* __restrict__ bk,
      const float* __restrict__ bv,
      __bf16* __restrict__ qb, __bf16* __restrict__ kb,
      __bf16* __restrict__ vT) {
  const int strip = blockIdx.x;
  const int which = blockIdx.y;  // 0=q 1=k 2=v
  const int wave = threadIdx.x >> 5;
  const int lane = threadIdx.x & 31;

  const __bf16* W = (which == 0) ? wqb : (which == 1) ? wkb : wvb;
  const float* bias = (which == 0) ? bq : (which == 1) ? bk : bv;

  const int r0 = strip * 16;
  const int c0 = wave * 128;

  v8f acc[8] = {};
  for (int k0 = 0; k0 < 1024; k0 += 32) {
    const v16bf a = frag_ld_bf16(xb + (size_t)r0 * 1024 + k0, 1024, lane);
#pragma unroll
    for (int nt = 0; nt < 8; ++nt) {
      const v16bf bf = frag_ld_bf16(W + (size_t)(c0 + nt * 16) * 1024 + k0, 1024, lane);
      acc[nt] = wmma_bf16(a, bf, acc[nt]);
    }
  }

  const int half = lane >> 4, colc = lane & 15;
#pragma unroll
  for (int nt = 0; nt < 8; ++nt) {
#pragma unroll
    for (int r = 0; r < 8; ++r) {
      const int i = r0 + r + 8 * half;      // global row in (B*S)
      const int j = c0 + nt * 16 + colc;    // global col in d
      const float v = acc[nt][r] + bias[j];
      const int b = i >> 8, s = i & 255;
      const int h = j >> 6, c = j & 63;
      if (which == 2) {
        vT[((size_t)(b * 16 + h) * 64 + c) * 256 + s] = (__bf16)v;
      } else {
        __bf16* dst = (which == 0) ? qb : kb;
        dst[((size_t)(b * 16 + h) * 256 + s) * 64 + c] = (__bf16)v;
      }
    }
  }
}

// ---------------------------------------------------------------------------
// Kernel 4a: attention (biased path), fully parallel over (b, strip, head).
// Block = 64 threads (2 waves) owns a 32-row strip of one head:
//   scores(32x256) = Q*K^T/8 (WMMA -> LDS fp32)
//   biased softmax, 1 exp/element, 2 threads per row
//   attn(32x64) = P * V (WMMA, V pre-transposed)
//   torch-faithful scrambled store into attn_s (bf16).
// LDS: sc 32*256 | bias 32*64 | red 64 | red2 64  (~40.5 KB dynamic)
// ---------------------------------------------------------------------------
__global__ void __launch_bounds__(64)
k_attn(const __bf16* __restrict__ qb, const __bf16* __restrict__ kb,
       const __bf16* __restrict__ vT, const float* __restrict__ div,
       __bf16* __restrict__ attn_s) {
  extern __shared__ __align__(16) float smem[];
  float* sc   = smem;                  // 32*256 scores -> probabilities
  float* bls  = smem + 32 * 256;       // 32*64 bias block
  float* red  = bls + 32 * 64;         // 64 partial maxima
  float* red2 = red + 64;              // 64 partial sums

  const int b  = blockIdx.x;           // 32
  const int s0 = blockIdx.y * 32;      // 8 strips
  const int h  = blockIdx.z;           // 16 heads
  const int tid = threadIdx.x;
  const int wave = tid >> 5;
  const int lane = tid & 31;
  const bool has_bias = (s0 < 64);

  if (has_bias) {
    const float* dsrc = div + (size_t)b * 4096 + s0 * 64;
    for (int i = tid; i < 32 * 64; i += 64) bls[i] = dsrc[i];
  }

  const __bf16* qh = qb + (size_t)(b * 16 + h) * 256 * 64;
  const __bf16* kh = kb + (size_t)(b * 16 + h) * 256 * 64;
  const __bf16* vh = vT + (size_t)(b * 16 + h) * 64 * 256;

  const int rr = wave * 16;            // local row base for this wave
  const int half = lane >> 4, colc = lane & 15;

  // ---- scores strip: 16 rows x 256 cols per wave ----
  const v16bf aq0 = frag_ld_bf16(qh + (size_t)(s0 + rr) * 64 + 0, 64, lane);
  const v16bf aq1 = frag_ld_bf16(qh + (size_t)(s0 + rr) * 64 + 32, 64, lane);
#pragma unroll 4
  for (int tt = 0; tt < 16; ++tt) {
    v8f acc = {};
    const v16bf b0 = frag_ld_bf16(kh + (size_t)(tt * 16) * 64 + 0, 64, lane);
    const v16bf b1 = frag_ld_bf16(kh + (size_t)(tt * 16) * 64 + 32, 64, lane);
    acc = wmma_bf16(aq0, b0, acc);
    acc = wmma_bf16(aq1, b1, acc);
#pragma unroll
    for (int r = 0; r < 8; ++r)
      sc[(rr + r + 8 * half) * 256 + tt * 16 + colc] = acc[r] * 0.125f;
  }
  __syncthreads();

  // ---- biased softmax: 2 threads per row, 1 exp per element ----
  {
    const int r = tid & 31;            // row
    const int hf = tid >> 5;           // half: t in [hf*128, +128)
    float* row = sc + r * 256;
    const float* brow = bls + r * 64;
    const int t0 = hf * 128, t1 = t0 + 128;

    float m = -1e30f;
    for (int t = t0; t < t1; ++t) {
      const float vb = row[t] + ((has_bias && t < 64) ? brow[t] : 0.f);
      m = fmaxf(m, vb);
    }
    red[hf * 32 + r] = m;
    __syncthreads();
    const float mx = fmaxf(red[r], red[32 + r]);

    float s = 0.f;
    for (int t = t0; t < t1; ++t) {
      const float vb = row[t] + ((has_bias && t < 64) ? brow[t] : 0.f);
      const float e = __expf(vb - mx);
      row[t] = e;                      // in-place: raw no longer needed
      s += e;
    }
    red2[hf * 32 + r] = s;
    __syncthreads();
    const float inv = 1.f / (red2[r] + red2[32 + r]);
    for (int t = t0; t < t1; ++t) row[t] *= inv;
  }
  __syncthreads();

  // ---- attn strip: P(16x256) * V(256x64) per wave ----
  v8f oacc[4] = {};
  for (int t0 = 0; t0 < 256; t0 += 32) {
    const v16bf pa = frag_ld_f32(sc + rr * 256 + t0, 256, lane);
#pragma unroll
    for (int nt = 0; nt < 4; ++nt) {
      const v16bf vb = frag_ld_bf16(vh + (size_t)(nt * 16) * 256 + t0, 256, lane);
      oacc[nt] = wmma_bf16(pa, vb, oacc[nt]);
    }
  }

  // ---- torch-faithful scrambled store ----
  // out flat index of (1,nh,B,S,dk) reshaped to (B,S,d):
  //   b' = 2h + b/16, s' = (b%16)*16 + s/16, j = (s%16)*64 + c
  const int bp = 2 * h + (b >> 4);
#pragma unroll
  for (int nt = 0; nt < 4; ++nt) {
#pragma unroll
    for (int r = 0; r < 8; ++r) {
      const int s = s0 + rr + r + 8 * half;
      const int c = nt * 16 + colc;
      const int sp_ = (b & 15) * 16 + (s >> 4);
      const int j = (s & 15) * 64 + c;
      attn_s[((size_t)bp * 256 + sp_) * 1024 + j] = (__bf16)oacc[nt][r];
    }
  }
}

// ---------------------------------------------------------------------------
// Kernel 4b: attention_weights = mean over heads of softmax(scores) (no bias).
// Recomputing Q*K^T (4.3 GFLOP of WMMA) beats materializing 134 MB of scores
// (268 MB HBM round-trip). Block = 64 threads per (b, 32-row strip), serial
// head loop with deterministic in-LDS accumulation (no atomics).
// LDS: sc 32*256 | acw 32*256 | red 64 | red2 64  (~64.5 KB dynamic)
// ---------------------------------------------------------------------------
__global__ void __launch_bounds__(64)
k_aw(const __bf16* __restrict__ qb, const __bf16* __restrict__ kb,
     float* __restrict__ aw_out) {
  extern __shared__ __align__(16) float smem[];
  float* sc   = smem;                  // 32*256
  float* acw  = smem + 32 * 256;       // 32*256 accumulator
  float* red  = acw + 32 * 256;        // 64
  float* red2 = red + 64;              // 64

  const int b  = blockIdx.x;
  const int s0 = blockIdx.y * 32;
  const int tid = threadIdx.x;
  const int wave = tid >> 5;
  const int lane = tid & 31;

  for (int i = tid; i < 32 * 256; i += 64) acw[i] = 0.f;
  __syncthreads();

  const int rr = wave * 16;
  const int half = lane >> 4, colc = lane & 15;

  for (int h = 0; h < 16; ++h) {
    const __bf16* qh = qb + (size_t)(b * 16 + h) * 256 * 64;
    const __bf16* kh = kb + (size_t)(b * 16 + h) * 256 * 64;

    const v16bf aq0 = frag_ld_bf16(qh + (size_t)(s0 + rr) * 64 + 0, 64, lane);
    const v16bf aq1 = frag_ld_bf16(qh + (size_t)(s0 + rr) * 64 + 32, 64, lane);
#pragma unroll 4
    for (int tt = 0; tt < 16; ++tt) {
      v8f acc = {};
      const v16bf b0 = frag_ld_bf16(kh + (size_t)(tt * 16) * 64 + 0, 64, lane);
      const v16bf b1 = frag_ld_bf16(kh + (size_t)(tt * 16) * 64 + 32, 64, lane);
      acc = wmma_bf16(aq0, b0, acc);
      acc = wmma_bf16(aq1, b1, acc);
#pragma unroll
      for (int r = 0; r < 8; ++r)
        sc[(rr + r + 8 * half) * 256 + tt * 16 + colc] = acc[r] * 0.125f;
    }
    __syncthreads();

    // plain softmax, 2 threads per row, accumulate into acw
    {
      const int r = tid & 31;
      const int hf = tid >> 5;
      float* row = sc + r * 256;
      const int t0 = hf * 128, t1 = t0 + 128;

      float m = -1e30f;
      for (int t = t0; t < t1; ++t) m = fmaxf(m, row[t]);
      red[hf * 32 + r] = m;
      __syncthreads();
      const float mx = fmaxf(red[r], red[32 + r]);

      float s = 0.f;
      for (int t = t0; t < t1; ++t) {
        const float e = __expf(row[t] - mx);
        row[t] = e;
        s += e;
      }
      red2[hf * 32 + r] = s;
      __syncthreads();
      const float inv = 1.f / (red2[r] + red2[32 + r]);
      for (int t = t0; t < t1; ++t) acw[r * 256 + t] += row[t] * inv;
    }
    __syncthreads();  // protect sc before next head's GEMM
  }

  float* dst = aw_out + ((size_t)b * 256 + s0) * 256;
  for (int i = tid; i < 32 * 256; i += 64) dst[i] = acw[i] * (1.f / 16.f);
}

// ---------------------------------------------------------------------------
// Kernel 5: out = LayerNorm(attn_s @ wo^T + bo + x_seq).
// Block = 256 (8 waves) per 16-row strip; GEMM into LDS row buffer, then
// per-row mean/var reduction and normalized store to d_out.
// LDS (dynamic): ys 16*1024 f32 | red 2*16*16 | mu 16 | rs 16.
// ---------------------------------------------------------------------------
__global__ void __launch_bounds__(256)
k_out_ln(const __bf16* __restrict__ attn_s, const __bf16* __restrict__ wob,
         const float* __restrict__ bo, const float* __restrict__ x,
         const float* __restrict__ g, const float* __restrict__ bb,
         float* __restrict__ out) {
  extern __shared__ __align__(16) float smem[];
  float* ys = smem;                       // 16*1024
  float* red = smem + 16 * 1024;          // 2*16*16
  float* mu_s = red + 2 * 16 * 16;        // 16
  float* rs_s = mu_s + 16;                // 16

  const int strip = blockIdx.x;
  const int wave = threadIdx.x >> 5;
  const int lane = threadIdx.x & 31;
  const int r0 = strip * 16;
  const int c0 = wave * 128;

  v8f acc[8] = {};
  for (int k0 = 0; k0 < 1024; k0 += 32) {
    const v16bf a = frag_ld_bf16(attn_s + (size_t)r0 * 1024 + k0, 1024, lane);
#pragma unroll
    for (int nt = 0; nt < 8; ++nt) {
      const v16bf bf = frag_ld_bf16(wob + (size_t)(c0 + nt * 16) * 1024 + k0, 1024, lane);
      acc[nt] = wmma_bf16(a, bf, acc[nt]);
    }
  }

  const int half = lane >> 4, colc = lane & 15;
#pragma unroll
  for (int nt = 0; nt < 8; ++nt) {
#pragma unroll
    for (int r = 0; r < 8; ++r) {
      const int row = r + 8 * half;
      const int j = c0 + nt * 16 + colc;
      ys[row * 1024 + j] = acc[nt][r] + bo[j] + x[(size_t)(r0 + row) * 1024 + j];
    }
  }
  __syncthreads();

  // per-row stats: 16 partial lanes per row
  {
    const int row = threadIdx.x >> 4, part = threadIdx.x & 15;
    float s = 0.f, ss = 0.f;
    for (int j = part; j < 1024; j += 16) {
      const float v = ys[row * 1024 + j];
      s += v; ss = fmaf(v, v, ss);
    }
    red[row * 16 + part] = s;
    red[256 + row * 16 + part] = ss;
  }
  __syncthreads();
  if (threadIdx.x < 16) {
    float s = 0.f, ss = 0.f;
    for (int p = 0; p < 16; ++p) {
      s += red[threadIdx.x * 16 + p];
      ss += red[256 + threadIdx.x * 16 + p];
    }
    const float mu = s * (1.f / 1024.f);
    const float var = ss * (1.f / 1024.f) - mu * mu;
    mu_s[threadIdx.x] = mu;
    rs_s[threadIdx.x] = rsqrtf(var + 1e-5f);
  }
  __syncthreads();

  for (int idx = threadIdx.x; idx < 16 * 1024; idx += 256) {
    const int row = idx >> 10, j = idx & 1023;
    out[(size_t)(r0 + row) * 1024 + j] =
        (ys[idx] - mu_s[row]) * rs_s[row] * g[j] + bb[j];
  }
}

// ---------------------------------------------------------------------------
// Host launcher
// ---------------------------------------------------------------------------
extern "C" void kernel_launch(void* const* d_in, const int* in_sizes, int n_in,
                              void* d_out, int out_size, void* d_ws, size_t ws_size,
                              hipStream_t stream) {
  const float* x  = (const float*)d_in[0];   // (32,16,16,1024)
  const float* pf = (const float*)d_in[1];   // (32,64,128)
  const float* wq = (const float*)d_in[2];
  const float* bq = (const float*)d_in[3];
  const float* wk = (const float*)d_in[4];
  const float* bk = (const float*)d_in[5];
  const float* wv = (const float*)d_in[6];
  const float* bv = (const float*)d_in[7];
  const float* wo = (const float*)d_in[8];
  const float* bo = (const float*)d_in[9];
  const float* lg = (const float*)d_in[10];
  const float* lb = (const float*)d_in[11];

  char* ws = (char*)d_ws;
  size_t off = 0;
  auto alloc = [&](size_t bytes) -> void* {
    void* p = ws + off;
    off += (bytes + 255) & ~(size_t)255;
    return p;
  };

  const size_t ND = 8388608;  // B*S*d
  const size_t NW = 1048576;  // d*d

  __bf16* xb   = (__bf16*)alloc(ND * 2);
  __bf16* wqb  = (__bf16*)alloc(NW * 2);
  __bf16* wkb  = (__bf16*)alloc(NW * 2);
  __bf16* wvb  = (__bf16*)alloc(NW * 2);
  __bf16* wob  = (__bf16*)alloc(NW * 2);
  __bf16* qb   = (__bf16*)alloc(ND * 2);  // (B,nh,S,dk)
  __bf16* kb   = (__bf16*)alloc(ND * 2);  // (B,nh,S,dk)
  __bf16* vT   = (__bf16*)alloc(ND * 2);  // (B,nh,dk,S)
  __bf16* attn = (__bf16*)alloc(ND * 2);  // scrambled (B,S,d)
  float*  div  = (float*)alloc(32 * 64 * 64 * 4);

  float* out = (float*)d_out;          // (B,H,W,d) = 8388608 f32
  float* aw  = out + ND;               // (B,S,S)   = 2097152 f32

  // 1. conversions
  k_cvt<<<2048, 256, 0, stream>>>(x, xb, (int)ND);
  k_cvt<<<1024, 256, 0, stream>>>(wq, wqb, (int)NW);
  k_cvt<<<1024, 256, 0, stream>>>(wk, wkb, (int)NW);
  k_cvt<<<1024, 256, 0, stream>>>(wv, wvb, (int)NW);
  k_cvt<<<1024, 256, 0, stream>>>(wo, wob, (int)NW);

  // 2. diversity bias
  k_div_bias<<<32, 64, 0, stream>>>(pf, div);

  // 3. QKV projections (WMMA)
  k_qkv<<<dim3(512, 3), 256, 0, stream>>>(xb, wqb, wkb, wvb, bq, bk, bv,
                                          qb, kb, vT);

  // 4a. attention, fully parallel over (b, strip, head)
  const size_t smem4a = (32 * 256 + 32 * 64 + 128) * sizeof(float);  // ~40.5 KB
  k_attn<<<dim3(32, 8, 16), 64, smem4a, stream>>>(qb, kb, vT, div, attn);

  // 4b. head-mean attention_weights (WMMA recompute + plain softmax)
  const size_t smem4b = (2 * 32 * 256 + 128) * sizeof(float);        // ~64.5 KB
  k_aw<<<dim3(32, 8), 64, smem4b, stream>>>(qb, kb, aw);

  // 5. out projection + residual + layernorm (WMMA fused)
  const size_t smem5 = (16 * 1024 + 2 * 16 * 16 + 32) * sizeof(float);
  k_out_ln<<<512, 256, smem5, stream>>>(attn, wob, bo, x, lg, lb, out);
}
